// TransformerEncoderBlock_47339129536738
// MI455X (gfx1250) — matrix operand order repaired and checked by reading, not compiled, not run on hardware
//
#include <hip/hip_runtime.h>
#include <hip/hip_bf16.h>
#include <math.h>
#include <stdint.h>

// ---------------------------------------------------------------------------
// CDNA5 / gfx1250, wave32.  All GEMMs: double-buffered async-LDS pipeline
// (global_load_async_to_lds_b128 -> ds_load_b128 -> v_wmma_f32_16x16x32_bf16).
// Shapes: B=16 S=1024 D=768 H=12 DH=64 F=3072 -> 284 GFLOP of GEMM.
// ---------------------------------------------------------------------------

typedef __bf16 bf16;
typedef __attribute__((ext_vector_type(16))) bf16  v16bf;
typedef __attribute__((ext_vector_type(8)))  float v8f;
typedef __attribute__((ext_vector_type(4)))  float f32x4;
typedef __attribute__((ext_vector_type(4)))  int   v4i;
typedef __attribute__((address_space(1))) v4i* as1_v4i;
typedef __attribute__((address_space(3))) v4i* as3_v4i;

static constexpr int D_MODEL = 768;
static constexpr int NHEAD   = 12;
static constexpr int DHEAD   = 64;
static constexpr int SEQ     = 1024;
static constexpr int BATCH   = 16;
static constexpr int FFN     = 3072;
static constexpr long NTOK   = (long)BATCH * SEQ;   // 16384
static constexpr int LDP     = 40;                  // LDS row pitch (bf16): 80B, 16B-aligned, conflict-benign

__device__ __forceinline__ bf16 f2bf(float f) {
    unsigned u = __float_as_uint(f);
    unsigned r = (u + 0x7FFFu + ((u >> 16) & 1u)) >> 16;   // RNE
    unsigned short h = (unsigned short)r;
    bf16 b;
    __builtin_memcpy(&b, &h, 2);
    return b;
}

// ------------------- async global->LDS copy (ASYNCcnt path) ----------------
#if __has_builtin(__builtin_amdgcn_global_load_async_to_lds_b128)
#define ASYNC_LDS_BUILTIN 1
#endif

__device__ __forceinline__ void async_cp16(void* lds_dst, const void* gsrc) {
    // LDS byte offset = flat address [31:0] (LDS aperture mapping, ISA 10.2)
    const unsigned loff = (unsigned)(uintptr_t)lds_dst;
#ifdef ASYNC_LDS_BUILTIN
    __builtin_amdgcn_global_load_async_to_lds_b128(
        (as1_v4i)(uintptr_t)gsrc, (as3_v4i)(uintptr_t)loff, 0, 0);
#else
    asm volatile("global_load_async_to_lds_b128 %0, %1, off"
                 :: "v"(loff), "v"(gsrc) : "memory");
#endif
}

template <int N>
__device__ __forceinline__ void wait_async() {
#if __has_builtin(__builtin_amdgcn_s_wait_asynccnt)
    __builtin_amdgcn_s_wait_asynccnt(N);
#else
    asm volatile("s_wait_asynccnt %0" :: "i"(N) : "memory");
#endif
}

// ---------------- WMMA fragment loaders (contiguous-K order, same for A and B) -------------
// A 16x32 bf16: lane L holds row m=L%16, 16 contiguous K at half*16
__device__ __forceinline__ v16bf load_frag_a(const bf16* base, int ld) {
    const int lane = threadIdx.x & 31;
    const int r = lane & 15, half = lane >> 4;
    const bf16* p = base + (long)r * ld + half * 16;
    union { v16bf v; f32x4 q[2]; } f;
    f.q[0] = *reinterpret_cast<const f32x4*>(p);        // K = half*16 .. +7
    f.q[1] = *reinterpret_cast<const f32x4*>(p + 8);    // K = half*16+8 .. +15
    return f.v;
}
// B 32x16 bf16 where Wt is stored [N,K] (so B = Wt^T): lane L holds
// column n=L%16 (= row n of Wt), 16 contiguous K at half*16.
__device__ __forceinline__ v16bf load_frag_b(const bf16* base, int ld) {
    const int lane = threadIdx.x & 31;
    const int n = lane & 15, half = lane >> 4;
    const bf16* p = base + (long)n * ld + half * 16;
    union { v16bf v; f32x4 q[2]; } f;
    f.q[0] = *reinterpret_cast<const f32x4*>(p);
    f.q[1] = *reinterpret_cast<const f32x4*>(p + 8);
    return f.v;
}

// ------------------------------ GEMM kernel --------------------------------
enum { EPI_SCORES = 0, EPI_QKV = 1, EPI_CTX = 2, EPI_RES = 3, EPI_GELU = 4 };

struct GemmP {
    const bf16* A;  long sA; int lda;
    const bf16* W;  long sW; int ldw;
    float*      C;  long sC; int ldc;
    bf16*       Cb; int ldcb;
    const float* bias;
    const float* res; int ldres;
    bf16 *q, *k, *vT;
    float scale;
    int  K;
};

// 256 threads = 8 waves; wave tile 32x32 (2x2 wmma), block tile 128(M)x64(N).
// Per 32-K stage: A tile 128x32 (512 x 16B chunks), W tile 64x32 (256 chunks)
// -> each thread issues exactly 2 A-chunk + 1 W-chunk async copies (uniform
// per-wave ASYNCcnt), double buffered in LDS.
template <int EPI>
__global__ __launch_bounds__(256) void gemm_k(GemmP p) {
    __shared__ bf16 lA[2][128 * LDP];
    __shared__ bf16 lW[2][64 * LDP];

    const int t     = threadIdx.x;
    const int wid   = t >> 5;
    const int lane  = t & 31;
    const int waveM = wid >> 1, waveN = wid & 1;
    const int m0 = blockIdx.y * 128 + waveM * 32;
    const int n0 = blockIdx.x * 64  + waveN * 32;
    const int bz = blockIdx.z;

    const bf16* Ablk = p.A + (long)bz * p.sA + (long)(blockIdx.y * 128) * p.lda;
    const bf16* Wblk = p.W + (long)bz * p.sW + (long)(blockIdx.x * 64)  * p.ldw;

    // chunk decomposition (16B = 8 bf16)
    const int arow0 = t >> 2,        akq = (t & 3) * 8;        // A chunk 0: rows 0..63
    const int arow1 = (t + 256) >> 2;                          // A chunk 1: rows 64..127
    const int wrow  = t >> 2,        wkq = (t & 3) * 8;        // W chunk : rows 0..63

    auto copy_stage = [&](int ks, int bufi) {
        const int k0 = ks * 32;
        async_cp16(&lA[bufi][arow0 * LDP + akq], Ablk + (long)arow0 * p.lda + k0 + akq);
        async_cp16(&lA[bufi][arow1 * LDP + akq], Ablk + (long)arow1 * p.lda + k0 + akq);
        async_cp16(&lW[bufi][wrow  * LDP + wkq], Wblk + (long)wrow  * p.ldw + k0 + wkq);
    };

    v8f c00 = {}; v8f c01 = {}; v8f c10 = {}; v8f c11 = {};

    const int nk = p.K >> 5;
    copy_stage(0, 0);
    for (int i = 0; i < nk; ++i) {
        const int cur = i & 1;
        if (i + 1 < nk) { copy_stage(i + 1, cur ^ 1); wait_async<3>(); }
        else            { wait_async<0>(); }
        __syncthreads();                               // stage i resident in LDS

        const bf16* aBase = &lA[cur][(waveM * 32) * LDP];
        const bf16* wBase = &lW[cur][(waveN * 32) * LDP];
        v16bf a0 = load_frag_a(aBase,            LDP);
        v16bf a1 = load_frag_a(aBase + 16 * LDP, LDP);
        v16bf b0 = load_frag_b(wBase,            LDP);
        v16bf b1 = load_frag_b(wBase + 16 * LDP, LDP);
        c00 = __builtin_amdgcn_wmma_f32_16x16x32_bf16(false, a0, false, b0, (short)0, c00, false, false);
        c01 = __builtin_amdgcn_wmma_f32_16x16x32_bf16(false, a0, false, b1, (short)0, c01, false, false);
        c10 = __builtin_amdgcn_wmma_f32_16x16x32_bf16(false, a1, false, b0, (short)0, c10, false, false);
        c11 = __builtin_amdgcn_wmma_f32_16x16x32_bf16(false, a1, false, b1, (short)0, c11, false, false);

        __syncthreads();                               // reads done before buffer reuse
    }

    // C/D layout: element (m,n) lives in vgpr v = m%8, lane = (m/8)*16 + n
    const int half = lane >> 4, nloc = lane & 15;
    v8f cc[2][2] = { { c00, c01 }, { c10, c11 } };
    #pragma unroll
    for (int mi = 0; mi < 2; ++mi)
    #pragma unroll
    for (int ni = 0; ni < 2; ++ni)
    #pragma unroll
    for (int v = 0; v < 8; ++v) {
        const int   mg  = m0 + mi * 16 + half * 8 + v;
        const int   ng  = n0 + ni * 16 + nloc;
        const float acc = cc[mi][ni][v];

        if (EPI == EPI_SCORES) {
            p.C[(long)bz * p.sC + (long)mg * p.ldc + ng] = acc * p.scale;
        } else if (EPI == EPI_QKV) {
            const float val = acc + p.bias[ng];
            const int sel = ng / D_MODEL;          // 0=q 1=k 2=v
            const int f   = ng - sel * D_MODEL;
            const int h   = f >> 6, d = f & 63;
            const int b   = mg >> 10, s = mg & 1023;
            const long bh = (long)b * NHEAD + h;
            const bf16 bv = f2bf(val);
            if (sel == 0)      p.q [(bh * SEQ + s) * DHEAD + d] = bv;
            else if (sel == 1) p.k [(bh * SEQ + s) * DHEAD + d] = bv;
            else               p.vT[(bh * DHEAD + d) * SEQ + s] = bv;   // V transposed
        } else if (EPI == EPI_CTX) {
            const int b = bz / NHEAD, h = bz % NHEAD;
            p.Cb[((long)b * SEQ + mg) * D_MODEL + h * DHEAD + ng] = f2bf(acc);
        } else if (EPI == EPI_RES) {
            p.C[(long)mg * p.ldc + ng] = acc + p.bias[ng] + p.res[(long)mg * p.ldres + ng];
        } else if (EPI == EPI_GELU) {
            const float tt = acc + p.bias[ng];
            const float gl = 0.5f * tt * (1.0f + erff(tt * 0.70710678118654752f));
            p.Cb[(long)mg * p.ldcb + ng] = f2bf(gl);
        }
    }
}

// ------------------------- block reductions (wave32) -----------------------
__device__ __forceinline__ float wave_max(float v) {
    #pragma unroll
    for (int o = 16; o > 0; o >>= 1) v = fmaxf(v, __shfl_xor(v, o, 32));
    return v;
}
__device__ __forceinline__ float wave_sum(float v) {
    #pragma unroll
    for (int o = 16; o > 0; o >>= 1) v += __shfl_xor(v, o, 32);
    return v;
}
__device__ __forceinline__ float block_max(float v, float* red) {
    const int lane = threadIdx.x & 31, wid = threadIdx.x >> 5;
    v = wave_max(v);
    __syncthreads();
    if (lane == 0) red[wid] = v;
    __syncthreads();
    float r = red[0];
    #pragma unroll
    for (int i = 1; i < 8; ++i) r = fmaxf(r, red[i]);
    return r;
}
__device__ __forceinline__ float block_sum(float v, float* red) {
    const int lane = threadIdx.x & 31, wid = threadIdx.x >> 5;
    v = wave_sum(v);
    __syncthreads();
    if (lane == 0) red[wid] = v;
    __syncthreads();
    float r = red[0];
    #pragma unroll
    for (int i = 1; i < 8; ++i) r += red[i];
    return r;
}

// ---------------- softmax over keys + mean over heads ----------------------
// One block per (b,q).  f32 score rows are rewritten IN PLACE as bf16
// probabilities (bf16 row occupies the first half of the 4KB f32 row, so the
// ctx GEMM reads them with ld = 2*SEQ).  Head-mean goes straight to d_out.
__global__ __launch_bounds__(256) void softmax_mean_k(float* scores, float* attn_mean) {
    const int b = blockIdx.x >> 10;
    const int q = blockIdx.x & 1023;
    const int tid = threadIdx.x;
    __shared__ float red[8];

    float macc[4] = {0.f, 0.f, 0.f, 0.f};
    for (int h = 0; h < NHEAD; ++h) {
        float* row = scores + (((long)b * NHEAD + h) * SEQ + q) * SEQ;
        float v0[4];
        #pragma unroll
        for (int j = 0; j < 4; ++j) v0[j] = row[tid + j * 256];
        float m = fmaxf(fmaxf(v0[0], v0[1]), fmaxf(v0[2], v0[3]));
        m = block_max(m, red);                 // barrier: all reads done before writes
        float e[4], s = 0.f;
        #pragma unroll
        for (int j = 0; j < 4; ++j) { e[j] = __expf(v0[j] - m); s += e[j]; }
        s = block_sum(s, red);
        const float inv = 1.0f / s;
        bf16* brow = reinterpret_cast<bf16*>(row);
        #pragma unroll
        for (int j = 0; j < 4; ++j) {
            const float pj = e[j] * inv;
            brow[tid + j * 256] = f2bf(pj);
            macc[j] += pj * (1.0f / NHEAD);
        }
        __syncthreads();
    }
    float* orow = attn_mean + ((long)b * SEQ + q) * SEQ;
    #pragma unroll
    for (int j = 0; j < 4; ++j) orow[tid + j * 256] = macc[j];
}

// --------------------------- layernorm -> bf16 -----------------------------
__global__ __launch_bounds__(256) void ln_k(const float* __restrict__ x,
                                            const float* __restrict__ g,
                                            const float* __restrict__ bb,
                                            bf16* __restrict__ out) {
    const long row = blockIdx.x;
    const float* xr = x + row * D_MODEL;
    const int tid = threadIdx.x;
    __shared__ float red[8];
    float v[3], s = 0.f, sq = 0.f;
    #pragma unroll
    for (int i = 0; i < 3; ++i) { v[i] = xr[tid + i * 256]; s += v[i]; sq += v[i] * v[i]; }
    s  = block_sum(s, red);
    sq = block_sum(sq, red);
    const float mean = s * (1.0f / D_MODEL);
    const float var  = sq * (1.0f / D_MODEL) - mean * mean;
    const float rs   = rsqrtf(var + 1e-5f);
    bf16* orow = out + row * D_MODEL;
    #pragma unroll
    for (int i = 0; i < 3; ++i) {
        const int idx = tid + i * 256;
        orow[idx] = f2bf((v[i] - mean) * rs * g[idx] + bb[idx]);
    }
}

// --------------------------- f32 -> bf16 copy ------------------------------
__global__ __launch_bounds__(256) void f32_to_bf16_k(const float* __restrict__ in,
                                                     bf16* __restrict__ out, long n) {
    for (long i = blockIdx.x * 256L + threadIdx.x; i < n; i += (long)gridDim.x * 256)
        out[i] = f2bf(in[i]);
}

// ---------------------------------------------------------------------------
extern "C" void kernel_launch(void* const* d_in, const int* in_sizes, int n_in,
                              void* d_out, int out_size, void* d_ws, size_t ws_size,
                              hipStream_t stream) {
    (void)in_sizes; (void)n_in; (void)out_size; (void)ws_size;
    const float* x     = (const float*)d_in[0];
    const float* ln1_g = (const float*)d_in[1];
    const float* ln1_b = (const float*)d_in[2];
    const float* w_in  = (const float*)d_in[3];
    const float* b_in  = (const float*)d_in[4];
    const float* w_out = (const float*)d_in[5];
    const float* b_out = (const float*)d_in[6];
    const float* ln2_g = (const float*)d_in[7];
    const float* ln2_b = (const float*)d_in[8];
    const float* w1    = (const float*)d_in[9];
    const float* b1    = (const float*)d_in[10];
    const float* w2    = (const float*)d_in[11];
    const float* b2    = (const float*)d_in[12];

    float* out_x    = (float*)d_out;                       // [B,S,D]
    float* out_attn = out_x + NTOK * D_MODEL;              // [B,S,S]

    // ----- workspace carve-out -----
    char*  base = (char*)d_ws;
    size_t off  = 0;
    auto alloc = [&](size_t bytes) -> char* {
        char* p = base + off; off += (bytes + 255) & ~(size_t)255; return p;
    };
    float* scores = (float*)alloc((size_t)BATCH * NHEAD * SEQ * SEQ * 4);  // 805 MB (reused as bf16 probs)
    bf16*  xn1    = (bf16*) alloc((size_t)NTOK * D_MODEL * 2);
    bf16*  qb     = (bf16*) alloc((size_t)BATCH * NHEAD * SEQ * DHEAD * 2);
    bf16*  kb     = (bf16*) alloc((size_t)BATCH * NHEAD * SEQ * DHEAD * 2);
    bf16*  vT     = (bf16*) alloc((size_t)BATCH * NHEAD * DHEAD * SEQ * 2);
    bf16*  ctx    = (bf16*) alloc((size_t)NTOK * D_MODEL * 2);
    float* x_res  = (float*)alloc((size_t)NTOK * D_MODEL * 4);
    bf16*  xn2    = (bf16*) alloc((size_t)NTOK * D_MODEL * 2);
    bf16*  h1     = (bf16*) alloc((size_t)NTOK * FFN * 2);
    bf16*  w_in_b = (bf16*) alloc((size_t)3 * D_MODEL * D_MODEL * 2);
    bf16*  w_out_b= (bf16*) alloc((size_t)D_MODEL * D_MODEL * 2);
    bf16*  w1_b   = (bf16*) alloc((size_t)FFN * D_MODEL * 2);
    bf16*  w2_b   = (bf16*) alloc((size_t)D_MODEL * FFN * 2);

    const dim3 blk(256);

    // weights -> bf16
    f32_to_bf16_k<<<dim3(1024), blk, 0, stream>>>(w_in,  w_in_b,  (long)3 * D_MODEL * D_MODEL);
    f32_to_bf16_k<<<dim3(1024), blk, 0, stream>>>(w_out, w_out_b, (long)D_MODEL * D_MODEL);
    f32_to_bf16_k<<<dim3(1024), blk, 0, stream>>>(w1,    w1_b,    (long)FFN * D_MODEL);
    f32_to_bf16_k<<<dim3(1024), blk, 0, stream>>>(w2,    w2_b,    (long)D_MODEL * FFN);

    // LN1
    ln_k<<<dim3((unsigned)NTOK), blk, 0, stream>>>(x, ln1_g, ln1_b, xn1);

    // QKV projection: [16384,768] x [2304,768]^T, scatter q/k/vT
    {
        GemmP p = {};
        p.A = xn1; p.sA = 0; p.lda = D_MODEL;
        p.W = w_in_b; p.sW = 0; p.ldw = D_MODEL;
        p.bias = b_in; p.q = qb; p.k = kb; p.vT = vT; p.K = D_MODEL;
        gemm_k<EPI_QKV><<<dim3(3 * D_MODEL / 64, NTOK / 128, 1), blk, 0, stream>>>(p);
    }
    // scores = (q k^T)/8 per (b,h): [1024,64] x [1024,64]^T
    {
        GemmP p = {};
        p.A = qb; p.sA = (long)SEQ * DHEAD; p.lda = DHEAD;
        p.W = kb; p.sW = (long)SEQ * DHEAD; p.ldw = DHEAD;
        p.C = scores; p.sC = (long)SEQ * SEQ; p.ldc = SEQ;
        p.scale = 0.125f; p.K = DHEAD;
        gemm_k<EPI_SCORES><<<dim3(SEQ / 64, SEQ / 128, BATCH * NHEAD), blk, 0, stream>>>(p);
    }
    // softmax (in-place f32 -> bf16) + head-mean into d_out
    softmax_mean_k<<<dim3((unsigned)(BATCH * SEQ)), blk, 0, stream>>>(scores, out_attn);

    // ctx = attn @ v per (b,h): [1024,1024(bf16, ld=2048)] x [64,1024]^T -> ctx[b,s,h*64+d]
    {
        GemmP p = {};
        p.A = (const bf16*)scores; p.sA = (long)2 * SEQ * SEQ; p.lda = 2 * SEQ;
        p.W = vT; p.sW = (long)DHEAD * SEQ; p.ldw = SEQ;
        p.Cb = ctx; p.K = SEQ;
        gemm_k<EPI_CTX><<<dim3(DHEAD / 64, SEQ / 128, BATCH * NHEAD), blk, 0, stream>>>(p);
    }
    // attn_out + x -> x_res : [16384,768] x [768,768]^T + b_out + x
    {
        GemmP p = {};
        p.A = ctx; p.lda = D_MODEL;
        p.W = w_out_b; p.ldw = D_MODEL;
        p.C = x_res; p.ldc = D_MODEL;
        p.bias = b_out; p.res = x; p.ldres = D_MODEL; p.K = D_MODEL;
        gemm_k<EPI_RES><<<dim3(D_MODEL / 64, NTOK / 128, 1), blk, 0, stream>>>(p);
    }
    // LN2
    ln_k<<<dim3((unsigned)NTOK), blk, 0, stream>>>(x_res, ln2_g, ln2_b, xn2);

    // h1 = gelu(xn2 @ w1^T + b1) : [16384,768] x [3072,768]^T
    {
        GemmP p = {};
        p.A = xn2; p.lda = D_MODEL;
        p.W = w1_b; p.ldw = D_MODEL;
        p.Cb = h1; p.ldcb = FFN;
        p.bias = b1; p.K = D_MODEL;
        gemm_k<EPI_GELU><<<dim3(FFN / 64, NTOK / 128, 1), blk, 0, stream>>>(p);
    }
    // out = h1 @ w2^T + b2 + x_res : [16384,3072] x [768,3072]^T
    {
        GemmP p = {};
        p.A = h1; p.lda = FFN;
        p.W = w2_b; p.ldw = FFN;
        p.C = out_x; p.ldc = D_MODEL;
        p.bias = b2; p.res = x_res; p.ldres = D_MODEL; p.K = FFN;
        gemm_k<EPI_RES><<<dim3(D_MODEL / 64, NTOK / 128, 1), blk, 0, stream>>>(p);
    }
}
